// Geo_FNO_2052994368051
// MI455X (gfx1250) — matrix operand order repaired and checked by reading, not compiled, not run on hardware
//
#include <hip/hip_runtime.h>
#include <hip/hip_bf16.h>
#include <math.h>

// ---------------- problem constants ----------------
#define BB   16
#define NN   4096
#define WW   32
#define M1C  12
#define M2C  12
#define S1C  40
#define S2C  40
#define MM1  24            // 2*M1
#define MM2  23            // 2*M2-1
#define KMV  552           // 24*23 valid modes
#define KMP  576           // padded to 18*32 for WMMA K-loop
#define HD   128           // 4*W iphi hidden
#define MTOT (BB*NN)       // 65536 rows

#define DIV_UP(a,b) (((a)+(b)-1)/(b))

typedef _Float16 half8  __attribute__((ext_vector_type(8)));
typedef _Float16 v16h   __attribute__((ext_vector_type(16)));
typedef float    v8f    __attribute__((ext_vector_type(8)));

// -------- WMMA fragment helpers (CDNA5 §7.12.2 layouts, wave32) --------
// A (16x32 f16, MxK): lanes0-15 = rows, K 0..7 & 16..23; lanes16-31 same rows, K 8..15 & 24..31
__device__ __forceinline__ v16h ldA(const _Float16* __restrict__ p, int ld) {
  int l = threadIdx.x & 31;
  const _Float16* q = p + (size_t)(l & 15) * ld + ((l >> 4) << 3);
  half8 lo = *(const half8*)q;
  half8 hi = *(const half8*)(q + 16);
  v16h r;
#pragma unroll
  for (int i = 0; i < 8; ++i) { r[i] = lo[i]; r[i + 8] = hi[i]; }
  return r;
}
// B (32x16 f16, KxN) loaded from B^T storage (row-major [col][k]):
// lanes0-15 = cols, K 0..7 & 16..23 ; lanes16-31 = cols, K 8..15 & 24..31
__device__ __forceinline__ v16h ldBT(const _Float16* p, int ld) {
  int l = threadIdx.x & 31;
  const _Float16* q = p + (size_t)(l & 15) * ld + ((l >> 4) << 3);
  half8 lo = *(const half8*)q;
  half8 hi = *(const half8*)(q + 16);
  v16h r;
#pragma unroll
  for (int i = 0; i < 8; ++i) { r[i] = lo[i]; r[i + 8] = hi[i]; }
  return r;
}
__device__ __forceinline__ v8f wmma_f16(v16h a, v16h b, v8f c) {
  return __builtin_amdgcn_wmma_f32_16x16x32_f16(false, a, false, b, (short)0, c,
                                                false, false);
}
__device__ __forceinline__ float gelu_exact(float x) {
  return 0.5f * x * (1.f + erff(x * 0.70710678118f));
}

// ---------------- small prep kernels ----------------
// w (rows x cols f32, [in][out]) -> wt f16 [out][in]
__global__ void k_transpose_f16(const float* __restrict__ w, _Float16* __restrict__ wt,
                                int rows, int cols) {
  int i = blockIdx.x * blockDim.x + threadIdx.x;
  if (i >= rows * cols) return;
  int o = i % cols, r = i / cols;
  wt[(size_t)o * rows + r] = (_Float16)w[(size_t)r * cols + o];
}

// cd[b][c] = code @ fc_code
__global__ void k_cd(const float* __restrict__ code, const float* __restrict__ w,
                     const float* __restrict__ b, float* __restrict__ cd) {
  int i = blockIdx.x * blockDim.x + threadIdx.x;
  if (i >= BB * WW) return;
  int c = i % WW, bi = i / WW;
  float s = b[c];
  for (int j = 0; j < 42; ++j) s += code[bi * 42 + j] * w[j * WW + c];
  cd[i] = s;
}

// iphi front end: build h row = [cd | xd@fc0+b | sin | cos] as f16 (Mtot x 128)
__global__ void k_iphi_front(const float* __restrict__ u, const float* __restrict__ cd,
                             const float* __restrict__ w0, const float* __restrict__ b0,
                             _Float16* __restrict__ h) {
  int i = blockIdx.x * blockDim.x + threadIdx.x;
  if (i >= MTOT) return;
  int b = i / NN;
  float x0 = u[(size_t)i * 2 + 0], x1 = u[(size_t)i * 2 + 1];
  float dx = x0 - 1e-4f, dy = x1 - 1e-4f;
  float xd[4] = {x0, x1, atan2f(dy, dx), sqrtf(dx * dx + dy * dy)};
  _Float16* row = h + (size_t)i * HD;
  const float* cdr = cd + b * WW;
#pragma unroll
  for (int c = 0; c < WW; ++c) row[c] = (_Float16)cdr[c];
  for (int c = 0; c < WW; ++c) {
    float s = b0[c];
#pragma unroll
    for (int d = 0; d < 4; ++d) s += xd[d] * w0[d * WW + c];
    row[WW + c] = (_Float16)s;
  }
#pragma unroll
  for (int d = 0; d < 4; ++d)
    for (int f = 0; f < 8; ++f) {
      float sn, cs;
      __sincosf(xd[d] * (3.14159265358979f * (float)(1 << f)), &sn, &cs);
      row[64 + d * 8 + f] = (_Float16)sn;
      row[96 + d * 8 + f] = (_Float16)cs;
    }
}

// ---------------- WMMA: 128x128 tanh MLP layer (weights staged in LDS) ----------------
__global__ void __launch_bounds__(256)
k_mlp128(const _Float16* __restrict__ hin, const _Float16* __restrict__ wt,
         const float* __restrict__ bias, _Float16* __restrict__ hout) {
  __shared__ _Float16 sW[HD * HD];  // 32 KB, B^T weights shared by all 8 waves
#pragma unroll
  for (int i = threadIdx.x * 8; i < HD * HD; i += 256 * 8)
    *(half8*)(sW + i) = *(const half8*)(wt + i);
  __syncthreads();

  int wave = (blockIdx.x * blockDim.x + threadIdx.x) >> 5;
  int m0 = wave * 16;
  if (m0 >= MTOT) return;
  int l = threadIdx.x & 31;
  v8f acc[8] = {};
  for (int k0 = 0; k0 < HD; k0 += 32) {
    v16h a = ldA(hin + (size_t)m0 * HD + k0, HD);
#pragma unroll
    for (int t = 0; t < 8; ++t) {
      v16h b = ldBT(sW + (size_t)(t * 16) * HD + k0, HD);  // ds_load path
      acc[t] = wmma_f16(a, b, acc[t]);
    }
  }
  int rl = (l >> 4) << 3, cl = l & 15;
#pragma unroll
  for (int t = 0; t < 8; ++t)
#pragma unroll
    for (int v = 0; v < 8; ++v) {
      int row = m0 + v + rl, c = t * 16 + cl;
      hout[(size_t)row * HD + c] = (_Float16)tanhf(acc[t][v] + bias[c]);
    }
}

// fc4 (128->2) + xi = x + x*h
__global__ void k_fc4_xi(const _Float16* __restrict__ h, const float* __restrict__ w,
                         const float* __restrict__ b, const float* __restrict__ u,
                         float* __restrict__ xi) {
  int i = blockIdx.x * blockDim.x + threadIdx.x;
  if (i >= MTOT) return;
  const _Float16* row = h + (size_t)i * HD;
  float s0 = b[0], s1 = b[1];
  for (int j = 0; j < HD; ++j) {
    float v = (float)row[j];
    s0 += v * w[j * 2 + 0];
    s1 += v * w[j * 2 + 1];
  }
  float x0 = u[(size_t)i * 2 + 0], x1 = u[(size_t)i * 2 + 1];
  xi[(size_t)i * 2 + 0] = x0 + x0 * s0;
  xi[(size_t)i * 2 + 1] = x1 + x1 * s1;
}

// uhT[b][c][n] = (u @ fc0 + b)^T  (f16, A-matrix of the encode GEMM)
__global__ void k_uh(const float* __restrict__ u, const float* __restrict__ w,
                     const float* __restrict__ b, _Float16* __restrict__ uhT) {
  int i = blockIdx.x * blockDim.x + threadIdx.x;
  if (i >= BB * WW * NN) return;
  int n = i % NN, c = (i / NN) % WW, bi = i / (NN * WW);
  float u0 = u[((size_t)bi * NN + n) * 2 + 0], u1 = u[((size_t)bi * NN + n) * 2 + 1];
  uhT[i] = (_Float16)(u0 * w[0 * WW + c] + u1 * w[1 * WW + c] + b[c]);
}

// NUFFT basis: cos/sin(2*pi*K) in two layouts (n-major for decode A, xy-major for encode B^T)
__global__ void k_basis(const float* __restrict__ xi,
                        _Float16* __restrict__ cN, _Float16* __restrict__ sN,
                        _Float16* __restrict__ cX, _Float16* __restrict__ sX) {
  size_t i = (size_t)blockIdx.x * blockDim.x + threadIdx.x;
  if (i >= (size_t)BB * NN * KMP) return;
  int xy = (int)(i % KMP);
  int n  = (int)((i / KMP) % NN);
  int b  = (int)(i / ((size_t)KMP * NN));
  float c = 0.f, s = 0.f;
  if (xy < KMV) {
    int kx = xy / MM2, ky = xy % MM2;
    float k1 = (kx < M1C) ? (float)kx : (float)(kx - MM1);
    float k2 = (ky < M2C) ? (float)ky : (float)(ky - MM2);
    float x0 = xi[((size_t)b * NN + n) * 2 + 0];
    float x1 = xi[((size_t)b * NN + n) * 2 + 1];
    __sincosf(6.28318530717959f * (x0 * k1 + x1 * k2), &s, &c);
  }
  cN[((size_t)b * NN + n) * KMP + xy] = (_Float16)c;
  sN[((size_t)b * NN + n) * KMP + xy] = (_Float16)s;
  cX[((size_t)b * KMP + xy) * NN + n] = (_Float16)c;
  sX[((size_t)b * KMP + xy) * NN + n] = (_Float16)s;
}

// ---------------- WMMA: NUFFT encode  u_ft[c,xy] = sum_n uh[c,n]*(cos - i sin) ----------------
__global__ void __launch_bounds__(256)
k_nufft_enc(const _Float16* __restrict__ uhT, const _Float16* __restrict__ cX,
            const _Float16* __restrict__ sX, float* __restrict__ ftRe,
            float* __restrict__ ftIm) {
  int wave = (blockIdx.x * blockDim.x + threadIdx.x) >> 5;
  int nt = wave % 36, mt = (wave / 36) & 1, b = wave / 72;
  if (b >= BB) return;
  int l = threadIdx.x & 31;
  v8f ar = {}, ai = {};
  const _Float16* Ab = uhT + ((size_t)b * WW + mt * 16) * NN;
  const _Float16* Cb = cX + ((size_t)b * KMP + nt * 16) * NN;
  const _Float16* Sb = sX + ((size_t)b * KMP + nt * 16) * NN;
  for (int k0 = 0; k0 < NN; k0 += 32) {
    v16h a  = ldA(Ab + k0, NN);
    v16h bc = ldBT(Cb + k0, NN);
    v16h bs = ldBT(Sb + k0, NN);
    ar = wmma_f16(a, bc, ar);
    ai = wmma_f16(a, bs, ai);
  }
  int rl = (l >> 4) << 3, col = nt * 16 + (l & 15);
#pragma unroll
  for (int v = 0; v < 8; ++v) {
    int c = mt * 16 + v + rl;
    size_t o = ((size_t)b * WW + c) * KMP + col;
    ftRe[o] = ar[v];
    ftIm[o] = -ai[v];   // exp(-2*pi*i*K)
  }
}

// ---------------- mode-space complex matmul (conv*_w1 / conv*_w2) ----------------
__global__ void k_modemm(const float* __restrict__ ftRe, const float* __restrict__ ftIm,
                         int ftRow, int ftLD, const float* __restrict__ w1,
                         const float* __restrict__ w2, float* __restrict__ fRe,
                         float* __restrict__ fIm) {
  int i = blockIdx.x * blockDim.x + threadIdx.x;
  if (i >= BB * WW * MM1 * M2C) return;
  int ky = i % M2C, kx = (i / M2C) % MM1, o = (i / (M2C * MM1)) % WW,
      b = i / (M2C * MM1 * WW);
  const float* w = (kx < M1C) ? w1 : w2;
  int kxw = (kx < M1C) ? kx : kx - M1C;
  float aR = 0.f, aI = 0.f;
  for (int ic = 0; ic < WW; ++ic) {
    size_t fo = ((size_t)b * WW + ic) * ftLD + (size_t)kx * ftRow + ky;
    float xr = ftRe[fo], xi = ftIm[fo];
    size_t wo = (((size_t)(ic * WW + o)) * M1C + kxw) * M2C + ky;
    float wr = w[2 * wo], wi = w[2 * wo + 1];
    aR += xr * wr - xi * wi;
    aI += xr * wi + xi * wr;
  }
  size_t oo = (((size_t)b * WW + o) * MM1 + kx) * M2C + ky;
  fRe[oo] = aR;
  fIm[oo] = aI;
}

// ifft along axis0 (rows -> 40 spatial), scale 1/40: g (B,32,40,12)
__global__ void k_ifft0(const float* __restrict__ fRe, const float* __restrict__ fIm,
                        float* __restrict__ gRe, float* __restrict__ gIm) {
  int i = blockIdx.x * blockDim.x + threadIdx.x;
  if (i >= BB * WW * S1C * M2C) return;
  int ky = i % M2C, s1 = (i / M2C) % S1C, c = (i / (M2C * S1C)) % WW,
      b = i / (M2C * S1C * WW);
  float aR = 0.f, aI = 0.f;
  for (int kx = 0; kx < MM1; ++kx) {
    int kxr = (kx < M1C) ? kx : kx - MM1;
    float sn, cs;
    __sincosf(6.28318530717959f * (float)kxr * (float)s1 / 40.f, &sn, &cs);
    size_t fo = (((size_t)b * WW + c) * MM1 + kx) * M2C + ky;
    float r = fRe[fo], im = fIm[fo];
    aR += r * cs - im * sn;
    aI += r * sn + im * cs;
  }
  size_t go = (((size_t)b * WW + c) * S1C + s1) * M2C + ky;
  gRe[go] = aR * 0.025f;
  gIm[go] = aI * 0.025f;
}

// irfft axis1 + optional pointwise conv (w) + grid bias + GELU
__global__ void k_combine(const float* __restrict__ gRe, const float* __restrict__ gIm,
                          const float* __restrict__ ucIn, const float* __restrict__ pw,
                          const float* __restrict__ bw, const float* __restrict__ bb,
                          float* __restrict__ ucOut) {
  int i = blockIdx.x * blockDim.x + threadIdx.x;
  if (i >= BB * WW * S1C * S2C) return;
  int s2 = i % S2C, s1 = (i / S2C) % S1C, o = (i / (S2C * S1C)) % WW,
      b = i / (S2C * S1C * WW);
  size_t gb = (((size_t)b * WW + o) * S1C + s1) * M2C;
  float acc = gRe[gb];  // ky = 0
  for (int ky = 1; ky < M2C; ++ky) {
    float sn, cs;
    __sincosf(6.28318530717959f * (float)ky * (float)s2 / 40.f, &sn, &cs);
    acc += 2.f * (gRe[gb + ky] * cs - gIm[gb + ky] * sn);
  }
  acc *= 0.025f;
  acc += bw[o * 2 + 0] * ((float)s1 / 39.f) + bw[o * 2 + 1] * ((float)s2 / 39.f) + bb[o];
  if (ucIn) {
    size_t pb = (size_t)b * WW * (S1C * S2C) + (size_t)s1 * S2C + s2;
    float s = 0.f;
    for (int ci = 0; ci < WW; ++ci) s += pw[o * WW + ci] * ucIn[pb + (size_t)ci * (S1C * S2C)];
    acc += s;
  }
  ucOut[(((size_t)b * WW + o) * S1C + s1) * S2C + s2] = gelu_exact(acc);
}

// rfft2 step 1: t[b,c,kx,s2] = sum_s1 uc * e^{-2*pi*i*kxr*s1/40}
__global__ void k_rfft1(const float* __restrict__ uc, float* __restrict__ tRe,
                        float* __restrict__ tIm) {
  int i = blockIdx.x * blockDim.x + threadIdx.x;
  if (i >= BB * WW * MM1 * S2C) return;
  int s2 = i % S2C, kx = (i / S2C) % MM1, c = (i / (S2C * MM1)) % WW,
      b = i / (S2C * MM1 * WW);
  int kxr = (kx < M1C) ? kx : kx - MM1;
  float aR = 0.f, aI = 0.f;
  for (int s1 = 0; s1 < S1C; ++s1) {
    float sn, cs;
    __sincosf(6.28318530717959f * (float)kxr * (float)s1 / 40.f, &sn, &cs);
    float v = uc[(((size_t)b * WW + c) * S1C + s1) * S2C + s2];
    aR += v * cs;
    aI -= v * sn;
  }
  size_t to = (((size_t)b * WW + c) * MM1 + kx) * S2C + s2;
  tRe[to] = aR;
  tIm[to] = aI;
}

// rfft2 step 2: ft[b,c,kx,ky] = sum_s2 t * e^{-2*pi*i*ky*s2/40}, ky<12
__global__ void k_rfft2(const float* __restrict__ tRe, const float* __restrict__ tIm,
                        float* __restrict__ rRe, float* __restrict__ rIm) {
  int i = blockIdx.x * blockDim.x + threadIdx.x;
  if (i >= BB * WW * MM1 * M2C) return;
  int ky = i % M2C, kx = (i / M2C) % MM1, c = (i / (M2C * MM1)) % WW,
      b = i / (M2C * MM1 * WW);
  float aR = 0.f, aI = 0.f;
  for (int s2 = 0; s2 < S2C; ++s2) {
    float sn, cs;
    __sincosf(6.28318530717959f * (float)ky * (float)s2 / 40.f, &sn, &cs);
    size_t to = (((size_t)b * WW + c) * MM1 + kx) * S2C + s2;
    float r = tRe[to], im = tIm[to];
    aR += r * cs + im * sn;
    aI += im * cs - r * sn;
  }
  size_t oo = (((size_t)b * WW + c) * MM1 + kx) * M2C + ky;
  rRe[oo] = aR;
  rIm[oo] = aI;
}

// hermitian completion -> decode B^T operands (Re, -Im) f16 (B,32,KMP)
__global__ void k_decprep(const float* __restrict__ fRe, const float* __restrict__ fIm,
                          _Float16* __restrict__ AuRe, _Float16* __restrict__ AuIm) {
  int i = blockIdx.x * blockDim.x + threadIdx.x;
  if (i >= BB * WW * KMP) return;
  int xy = i % KMP, c = (i / KMP) % WW, b = i / (KMP * WW);
  float r = 0.f, im = 0.f;
  if (xy < KMV) {
    int kx = xy / MM2, ky = xy % MM2;
    size_t bc = (size_t)b * WW + c;
    if (ky < M2C) {
      size_t o = (bc * MM1 + kx) * M2C + ky;
      r = fRe[o];
      im = fIm[o];
    } else {  // u_full[kx, ky] = conj(out_ft[23-kx, 23-ky])
      size_t o = (bc * MM1 + (MM1 - 1 - kx)) * M2C + (MM2 - ky);
      r = fRe[o];
      im = -fIm[o];
    }
  }
  AuRe[i] = (_Float16)r;
  AuIm[i] = (_Float16)(-im);  // store -Im: Re(u*e^{+i}) = Re*cos + (-Im)*sin
}

// ---------------- WMMA: NUFFT decode (out^T[n,c]) + b4 term -> uoN f16 ----------------
// All 8 waves of a block share one batch b; AuRe/AuIm for that batch staged in LDS (72 KB).
__global__ void __launch_bounds__(256)
k_nufft_dec(const _Float16* __restrict__ cN, const _Float16* __restrict__ sN,
            const _Float16* __restrict__ AuRe, const _Float16* __restrict__ AuIm,
            const float* __restrict__ u, const float* __restrict__ b4w,
            const float* __restrict__ b4b, _Float16* __restrict__ uoN) {
  __shared__ _Float16 sBr[WW * KMP];
  __shared__ _Float16 sBi[WW * KMP];
  int wave = (blockIdx.x * blockDim.x + threadIdx.x) >> 5;
  int mt = wave & 255, b = wave >> 8;   // 8 waves/block, 256 mtiles/batch -> same b per block
  const _Float16* gBr = AuRe + (size_t)b * WW * KMP;
  const _Float16* gBi = AuIm + (size_t)b * WW * KMP;
#pragma unroll
  for (int i = threadIdx.x * 8; i < WW * KMP; i += 256 * 8) {
    *(half8*)(sBr + i) = *(const half8*)(gBr + i);
    *(half8*)(sBi + i) = *(const half8*)(gBi + i);
  }
  __syncthreads();

  int m0 = mt * 16, l = threadIdx.x & 31;
  v8f a0 = {}, a1 = {};
  const _Float16* Ac = cN + ((size_t)b * NN + m0) * KMP;
  const _Float16* As = sN + ((size_t)b * NN + m0) * KMP;
  for (int k0 = 0; k0 < KMP; k0 += 32) {
    __builtin_prefetch(Ac + k0 + 64, 0, 1);
    v16h ac  = ldA(Ac + k0, KMP);
    v16h as  = ldA(As + k0, KMP);
    v16h b0r = ldBT(sBr + k0, KMP);
    v16h b0i = ldBT(sBi + k0, KMP);
    v16h b1r = ldBT(sBr + 16 * KMP + k0, KMP);
    v16h b1i = ldBT(sBi + 16 * KMP + k0, KMP);
    a0 = wmma_f16(ac, b0r, a0);
    a0 = wmma_f16(as, b0i, a0);
    a1 = wmma_f16(ac, b1r, a1);
    a1 = wmma_f16(as, b1i, a1);
  }
  int rl = (l >> 4) << 3, cl = l & 15;
#pragma unroll
  for (int v = 0; v < 8; ++v) {
    int n = m0 + v + rl;
    float u0 = u[((size_t)b * NN + n) * 2 + 0], u1 = u[((size_t)b * NN + n) * 2 + 1];
    int c = cl;
    float val = a0[v] + b4w[c * 2 + 0] * u0 + b4w[c * 2 + 1] * u1 + b4b[c];
    uoN[((size_t)b * NN + n) * WW + c] = (_Float16)val;
    c = 16 + cl;
    val = a1[v] + b4w[c * 2 + 0] * u0 + b4w[c * 2 + 1] * u1 + b4b[c];
    uoN[((size_t)b * NN + n) * WW + c] = (_Float16)val;
  }
}

// ---------------- WMMA: fc1 (32 -> 128) + GELU (weights in LDS) ----------------
__global__ void __launch_bounds__(256)
k_fc1(const _Float16* __restrict__ uoN, const _Float16* __restrict__ wt,
      const float* __restrict__ bias, _Float16* __restrict__ hf) {
  __shared__ _Float16 sW[HD * WW];  // 8 KB
#pragma unroll
  for (int i = threadIdx.x * 8; i < HD * WW; i += 256 * 8)
    *(half8*)(sW + i) = *(const half8*)(wt + i);
  __syncthreads();

  int wave = (blockIdx.x * blockDim.x + threadIdx.x) >> 5;
  int m0 = wave * 16;
  if (m0 >= MTOT) return;
  int l = threadIdx.x & 31;
  v16h a = ldA(uoN + (size_t)m0 * WW, WW);
#pragma unroll
  for (int t = 0; t < 8; ++t) {
    v16h b = ldBT(sW + (size_t)(t * 16) * WW, WW);
    v8f acc = {};
    acc = wmma_f16(a, b, acc);
    int rl = (l >> 4) << 3, col = t * 16 + (l & 15);
#pragma unroll
    for (int v = 0; v < 8; ++v) {
      float x = acc[v] + bias[col];
      hf[(size_t)(m0 + v + rl) * HD + col] = (_Float16)gelu_exact(x);
    }
  }
}

// fc2 (128 -> 1)
__global__ void k_fc2(const _Float16* __restrict__ hf, const float* __restrict__ w,
                      const float* __restrict__ b, float* __restrict__ out) {
  int i = blockIdx.x * blockDim.x + threadIdx.x;
  if (i >= MTOT) return;
  float s = b[0];
  const _Float16* row = hf + (size_t)i * HD;
  for (int j = 0; j < HD; ++j) s += (float)row[j] * w[j];
  out[i] = s;
}

// ---------------- host orchestration ----------------
extern "C" void kernel_launch(void* const* d_in, const int* in_sizes, int n_in,
                              void* d_out, int out_size, void* d_ws, size_t ws_size,
                              hipStream_t stream) {
  (void)in_sizes; (void)n_in; (void)out_size; (void)ws_size;
  // pytree-flattened params (sorted dict keys):
  const float* code  = (const float*)d_in[0];
  const float* u     = (const float*)d_in[1];
  const float* b0w = (const float*)d_in[2],  *b0b = (const float*)d_in[3];
  const float* b1w = (const float*)d_in[4],  *b1b = (const float*)d_in[5];
  const float* b2w = (const float*)d_in[6],  *b2b = (const float*)d_in[7];
  const float* b3w = (const float*)d_in[8],  *b3b = (const float*)d_in[9];
  const float* b4w = (const float*)d_in[10], *b4b = (const float*)d_in[11];
  const float* cw[5][2] = {{(const float*)d_in[12], (const float*)d_in[13]},
                           {(const float*)d_in[14], (const float*)d_in[15]},
                           {(const float*)d_in[16], (const float*)d_in[17]},
                           {(const float*)d_in[18], (const float*)d_in[19]},
                           {(const float*)d_in[20], (const float*)d_in[21]}};
  const float* fc0w = (const float*)d_in[22], *fc0b = (const float*)d_in[23];
  const float* fc1w = (const float*)d_in[24], *fc1b = (const float*)d_in[25];
  const float* fc2w = (const float*)d_in[26], *fc2b = (const float*)d_in[27];
  const float* i0w = (const float*)d_in[28], *i0b = (const float*)d_in[29];
  const float* i1w = (const float*)d_in[30], *i1b = (const float*)d_in[31];
  const float* i2w = (const float*)d_in[32], *i2b = (const float*)d_in[33];
  const float* i3w = (const float*)d_in[34], *i3b = (const float*)d_in[35];
  const float* i4w = (const float*)d_in[36], *i4b = (const float*)d_in[37];
  const float* icw = (const float*)d_in[38], *icb = (const float*)d_in[39];
  const float* pw_[4] = {nullptr, (const float*)d_in[40], (const float*)d_in[42],
                         (const float*)d_in[44]};
  const float* pb_[4] = {nullptr, (const float*)d_in[41], (const float*)d_in[43],
                         (const float*)d_in[45]};
  const float* gw_[4] = {b0w, b1w, b2w, b3w};
  const float* gb_[4] = {b0b, b1b, b2b, b3b};

  size_t off = 0;
  char* base = (char*)d_ws;
  auto alloc = [&](size_t bytes) -> char* {
    char* p = base + off;
    off += (bytes + 255) & ~(size_t)255;
    return p;
  };
  _Float16* wtI1 = (_Float16*)alloc(HD * HD * 2);
  _Float16* wtI2 = (_Float16*)alloc(HD * HD * 2);
  _Float16* wtI3 = (_Float16*)alloc(HD * HD * 2);
  _Float16* wtF1 = (_Float16*)alloc(HD * WW * 2);
  float*    cd   = (float*)alloc(BB * WW * 4);
  _Float16* hA   = (_Float16*)alloc((size_t)MTOT * HD * 2);
  _Float16* hB   = (_Float16*)alloc((size_t)MTOT * HD * 2);
  float*    xi   = (float*)alloc((size_t)MTOT * 2 * 4);
  _Float16* uhT  = (_Float16*)alloc((size_t)BB * WW * NN * 2);
  _Float16* cN   = (_Float16*)alloc((size_t)BB * NN * KMP * 2);
  _Float16* sN   = (_Float16*)alloc((size_t)BB * NN * KMP * 2);
  _Float16* cX   = (_Float16*)alloc((size_t)BB * KMP * NN * 2);
  _Float16* sX   = (_Float16*)alloc((size_t)BB * KMP * NN * 2);
  float* ftRe = (float*)alloc((size_t)BB * WW * KMP * 4);
  float* ftIm = (float*)alloc((size_t)BB * WW * KMP * 4);
  float* fRe  = (float*)alloc((size_t)BB * WW * MM1 * M2C * 4);
  float* fIm  = (float*)alloc((size_t)BB * WW * MM1 * M2C * 4);
  float* gRe  = (float*)alloc((size_t)BB * WW * S1C * M2C * 4);
  float* gIm  = (float*)alloc((size_t)BB * WW * S1C * M2C * 4);
  float* ucA  = (float*)alloc((size_t)BB * WW * S1C * S2C * 4);
  float* ucB  = (float*)alloc((size_t)BB * WW * S1C * S2C * 4);
  float* tRe  = (float*)alloc((size_t)BB * WW * MM1 * S2C * 4);
  float* tIm  = (float*)alloc((size_t)BB * WW * MM1 * S2C * 4);
  float* rRe  = (float*)alloc((size_t)BB * WW * MM1 * M2C * 4);
  float* rIm  = (float*)alloc((size_t)BB * WW * MM1 * M2C * 4);
  _Float16* AuRe = (_Float16*)alloc((size_t)BB * WW * KMP * 2);
  _Float16* AuIm = (_Float16*)alloc((size_t)BB * WW * KMP * 2);
  _Float16* uoN  = (_Float16*)alloc((size_t)MTOT * WW * 2);
  _Float16* hf   = (_Float16*)alloc((size_t)MTOT * HD * 2);

  const int T = 256;
  auto G = [&](long long n) { return dim3((unsigned)DIV_UP(n, T)); };

  // weight prep
  k_transpose_f16<<<G(HD * HD), T, 0, stream>>>(i1w, wtI1, HD, HD);
  k_transpose_f16<<<G(HD * HD), T, 0, stream>>>(i2w, wtI2, HD, HD);
  k_transpose_f16<<<G(HD * HD), T, 0, stream>>>(i3w, wtI3, HD, HD);
  k_transpose_f16<<<G(WW * HD), T, 0, stream>>>(fc1w, wtF1, WW, HD);

  // iphi: features + 3 WMMA tanh layers + fc4 -> xi
  k_cd<<<G(BB * WW), T, 0, stream>>>(code, icw, icb, cd);
  k_iphi_front<<<G(MTOT), T, 0, stream>>>(u, cd, i0w, i0b, hA);
  k_mlp128<<<G((long long)MTOT / 16 * 32), T, 0, stream>>>(hA, wtI1, i1b, hB);
  k_mlp128<<<G((long long)MTOT / 16 * 32), T, 0, stream>>>(hB, wtI2, i2b, hA);
  k_mlp128<<<G((long long)MTOT / 16 * 32), T, 0, stream>>>(hA, wtI3, i3b, hB);
  k_fc4_xi<<<G(MTOT), T, 0, stream>>>(hB, i4w, i4b, u, xi);

  // lift + basis + WMMA NUFFT encode
  k_uh<<<G((long long)BB * WW * NN), T, 0, stream>>>(u, fc0w, fc0b, uhT);
  k_basis<<<G((long long)BB * NN * KMP), T, 0, stream>>>(xi, cN, sN, cX, sX);
  k_nufft_enc<<<G((long long)BB * 2 * 36 * 32), T, 0, stream>>>(uhT, cX, sX, ftRe, ftIm);

  // conv0: mode matmul -> irfft2 -> +grid bias -> gelu
  k_modemm<<<G(BB * WW * MM1 * M2C), T, 0, stream>>>(ftRe, ftIm, MM2, KMP,
                                                     cw[0][0], cw[0][1], fRe, fIm);
  k_ifft0<<<G(BB * WW * S1C * M2C), T, 0, stream>>>(fRe, fIm, gRe, gIm);
  k_combine<<<G((long long)BB * WW * S1C * S2C), T, 0, stream>>>(
      gRe, gIm, nullptr, nullptr, b0w, b0b, ucA);

  // three standard FNO layers (ping-pong ucA/ucB)
  float* cin = ucA; float* cout = ucB;
  for (int L = 1; L <= 3; ++L) {
    k_rfft1<<<G(BB * WW * MM1 * S2C), T, 0, stream>>>(cin, tRe, tIm);
    k_rfft2<<<G(BB * WW * MM1 * M2C), T, 0, stream>>>(tRe, tIm, rRe, rIm);
    k_modemm<<<G(BB * WW * MM1 * M2C), T, 0, stream>>>(rRe, rIm, M2C, MM1 * M2C,
                                                       cw[L][0], cw[L][1], fRe, fIm);
    k_ifft0<<<G(BB * WW * S1C * M2C), T, 0, stream>>>(fRe, fIm, gRe, gIm);
    k_combine<<<G((long long)BB * WW * S1C * S2C), T, 0, stream>>>(
        gRe, gIm, cin, pw_[L], gw_[L], gb_[L], cout);
    float* tmp = cin; cin = cout; cout = tmp;
  }
  // cin now holds final uc (after 3 swaps)

  // conv4 decode: rfft2 -> mode matmul -> hermitian completion -> WMMA inverse NUFFT
  k_rfft1<<<G(BB * WW * MM1 * S2C), T, 0, stream>>>(cin, tRe, tIm);
  k_rfft2<<<G(BB * WW * MM1 * M2C), T, 0, stream>>>(tRe, tIm, rRe, rIm);
  k_modemm<<<G(BB * WW * MM1 * M2C), T, 0, stream>>>(rRe, rIm, M2C, MM1 * M2C,
                                                     cw[4][0], cw[4][1], fRe, fIm);
  k_decprep<<<G(BB * WW * KMP), T, 0, stream>>>(fRe, fIm, AuRe, AuIm);
  k_nufft_dec<<<G((long long)BB * 256 * 32), T, 0, stream>>>(cN, sN, AuRe, AuIm, u,
                                                             b4w, b4b, uoN);

  // head: WMMA fc1 + gelu, then fc2
  k_fc1<<<G((long long)MTOT / 16 * 32), T, 0, stream>>>(uoN, wtF1, fc1b, hf);
  k_fc2<<<G(MTOT), T, 0, stream>>>(hf, fc2w, fc2b, (float*)d_out);
}
